// GAFD_ABCD_74208444940808
// MI455X (gfx1250) — matrix operand order repaired and checked by reading, not compiled, not run on hardware
//
#include <hip/hip_runtime.h>

// ---------------------------------------------------------------------------
// GAFD + ABCD distillation loss for MI455X (gfx1250, wave32, WMMA).
// conv1 (618 GFLOP) runs as implicit-GEMM on v_wmma_f32_16x16x32_bf16 with
// double-buffered LDS staging via global_load_async_to_lds_b128 (ASYNCcnt).
// ---------------------------------------------------------------------------

typedef __attribute__((ext_vector_type(16))) __bf16 v16bf;
typedef __attribute__((ext_vector_type(8)))  float  v8f;

__device__ __forceinline__ __bf16 f2bf(float f) {
  unsigned int u = __builtin_bit_cast(unsigned int, f);
  unsigned int r = (u + 0x7FFFu + ((u >> 16) & 1u)) >> 16;
  return __builtin_bit_cast(__bf16, (unsigned short)r);
}

__device__ __forceinline__ v8f wmma_bf16(v16bf a, v16bf b, v8f c) {
  // (neg_a, A, neg_b, B, c_mod, C, reuse_a, reuse_b)
  return __builtin_amdgcn_wmma_f32_16x16x32_bf16(false, a, false, b,
                                                 (short)0, c, false, false);
}

__device__ __forceinline__ unsigned hash_u(unsigned x) {
  x ^= x >> 16; x *= 0x7feb352dU;
  x ^= x >> 15; x *= 0x846ca68bU;
  x ^= x >> 16;
  return x;
}

// Problem constants
constexpr int Bn = 16, Ct = 1024, Cs = 512, Hh = 64, Wd = 64;
constexpr int C1 = 512;            // conv1 out channels
constexpr int Np = 256;            // anchors / proj dim
constexpr float TAU_INV = 10.0f;   // 1 / 0.1
constexpr int TILE = 3 * 66 * 32;  // LDS tile: [row 3][col 66][ci 32] bf16

// ---------------------------------------------------------------------------
// Kernel 0a: pack conv1_w [512][1024][3][3] f32 -> [tap][co][ci] bf16
// ---------------------------------------------------------------------------
__global__ void pack_w1(const float* __restrict__ w, __bf16* __restrict__ out) {
  int idx = blockIdx.x * 256 + threadIdx.x;     // 9*512*1024
  if (idx >= 9 * C1 * Ct) return;
  int ci   = idx & 1023;
  int co   = (idx >> 10) & 511;
  int kpos = idx >> 19;
  out[idx] = f2bf(w[(size_t)(co * Ct + ci) * 9 + kpos]);
}

// Kernel 0b: pack proj weight [K][256] f32 -> transposed [256][K] bf16
__global__ void pack_proj(const float* __restrict__ w, __bf16* __restrict__ out, int K) {
  int idx = blockIdx.x * 256 + threadIdx.x;     // 256*K
  if (idx >= Np * K) return;
  int k = idx % K, p = idx / K;
  out[idx] = f2bf(w[(size_t)k * Np + p]);
}

// Kernel 0c: teacher f32 NCHW -> bf16 channel-last [b][y][x][ci]
// (so a 32-ci block at one pixel is 64 contiguous bytes -> async-DMA friendly)
__global__ void pack_teacher(const float* __restrict__ T, __bf16* __restrict__ out) {
  size_t idx = (size_t)blockIdx.x * 256 + threadIdx.x;   // 16*64*64*1024
  int c = (int)(idx & 1023);
  size_t pix = idx >> 10;            // b*4096 + y*64 + x
  int x  = (int)(pix & 63);
  int yy = (int)((pix >> 6) & 63);
  int b  = (int)(pix >> 12);
  out[idx] = f2bf(T[(((size_t)b * Ct + c) * Hh + yy) * Wd + x]);
}

__global__ void init_scal(float* scal) { scal[threadIdx.x] = 0.f; }

// ---------------------------------------------------------------------------
// Async staging of one [3][66][32] bf16 tile from channel-last teacher.
// Valid cells only (halo stays pre-zeroed); 768 x b128 lane-transfers.
// ---------------------------------------------------------------------------
__device__ __forceinline__ void stage_async(const __bf16* __restrict__ tb, int y, int cb,
                                            __bf16* buf, int tid) {
  #pragma unroll
  for (int uu = 0; uu < 3; ++uu) {
    int u  = tid + uu * 256;          // 0..767
    int q  = u & 3;                   // 16B quarter of the 64B ci-block
    int xx = (u >> 2) & 63;           // image column
    int r  = u >> 8;                  // tile row 0..2
    int yy = y + r - 1;
    if ((unsigned)yy < 64u) {
      unsigned lds_addr = (unsigned)(uintptr_t)(buf + ((r * 66 + xx + 1) * 32 + q * 8));
      unsigned long long gaddr = (unsigned long long)(uintptr_t)
          (tb + (((size_t)yy * 64 + xx) * 1024 + cb + q * 8));
      asm volatile("global_load_async_to_lds_b128 %0, %1, off"
                   :: "v"(lds_addr), "v"(gaddr) : "memory");
    }
  }
}

__device__ __forceinline__ void wait_async0() {
  asm volatile("s_wait_asynccnt 0" ::: "memory");
}

// ---------------------------------------------------------------------------
// Kernel 1: fused conv1(3x3,1024->512)+ReLU + conv2(1x1,512->1)+softplus.
// Implicit GEMM, per block = one (batch,row); per wave 64co x 64px tile.
// LDS tiles double-buffered and filled by the global_load_async_to_lds path.
// ---------------------------------------------------------------------------
__global__ __launch_bounds__(256) void heatmap_conv(
    const __bf16* __restrict__ Tpk, const __bf16* __restrict__ w1p,
    const float* __restrict__ b1, const float* __restrict__ w2,
    const float* __restrict__ b2, float* __restrict__ heat) {
  __shared__ __bf16 lds_in[2 * TILE];      // double buffer
  __shared__ float  lds_acc[64];
  const int b = blockIdx.y, y = blockIdx.x;
  const int tid  = threadIdx.x;
  const int lane = tid & 31, wave = tid >> 5;
  const int ln    = lane & 15;
  const int koff  = (lane & 16) ? 8 : 0;   // A-frag K base (ISA 16b A layout)
  const int khalf = lane & 16;             // unused; B frag uses koff like A
  const __bf16* tb = Tpk + (size_t)b * 64 * 64 * 1024;

  // Zero both buffers once (covers halo cells never touched by the DMA).
  const __bf16 zb = f2bf(0.f);
  for (int t = tid; t < 2 * TILE; t += 256) lds_in[t] = zb;
  __syncthreads();

  v8f vz = {0.f,0.f,0.f,0.f,0.f,0.f,0.f,0.f};
  v8f acc[4][4];
  for (int i = 0; i < 4; ++i)
    for (int j = 0; j < 4; ++j) acc[i][j] = vz;

  // Prime buffer 0.
  stage_async(tb, y, 0, lds_in, tid);
  wait_async0();
  __syncthreads();

  int cur = 0;
  for (int cb = 0; cb < Ct; cb += 32) {
    __bf16* buf = lds_in + cur * TILE;
    // Prefetch next ci-block into the other buffer while we compute.
    if (cb + 32 < Ct)
      stage_async(tb, y, cb + 32, lds_in + (cur ^ 1) * TILE, tid);

    for (int ky = 0; ky < 3; ++ky) {
      for (int kx = 0; kx < 3; ++kx) {
        const int kpos = ky * 3 + kx;
        v16bf afr[4];
        #pragma unroll
        for (int i = 0; i < 4; ++i) {
          const __bf16* ap = w1p +
              ((size_t)kpos * C1 + wave * 64 + i * 16 + ln) * Ct + cb + koff;
          #pragma unroll
          for (int j = 0; j < 8; ++j) { afr[i][j] = ap[j]; afr[i][8 + j] = ap[16 + j]; }
        }
        #pragma unroll
        for (int j = 0; j < 4; ++j) {
          v16bf bfr;
          const __bf16* bp = &buf[(ky * 66 + j * 16 + ln + kx) * 32 + koff];
          #pragma unroll
          for (int e = 0; e < 8; ++e) { bfr[e] = bp[e]; bfr[8 + e] = bp[16 + e]; }
          #pragma unroll
          for (int i = 0; i < 4; ++i) acc[i][j] = wmma_bf16(afr[i], bfr, acc[i][j]);
        }
      }
    }
    wait_async0();     // prefetch landed
    __syncthreads();   // everyone done reading `buf`, next overwrite is safe
    cur ^= 1;
  }

  if (tid < 64) lds_acc[tid] = 0.f;
  __syncthreads();
  // ReLU then 1x1 conv reduce over co via LDS float atomics (ds_add_f32).
  for (int i = 0; i < 4; ++i) {
    for (int j = 0; j < 4; ++j) {
      #pragma unroll
      for (int r = 0; r < 8; ++r) {
        int co = wave * 64 + i * 16 + ((lane & 16) ? 8 : 0) + r;
        int p  = j * 16 + ln;
        float v = fmaxf(acc[i][j][r] + b1[co], 0.f);
        atomicAdd(&lds_acc[p], w2[co] * v);
      }
    }
  }
  __syncthreads();
  if (tid < 64) {
    float hv = lds_acc[tid] + b2[0];
    float sp = fmaxf(hv, 0.f) + log1pf(expf(-fabsf(hv)));  // stable softplus
    heat[((size_t)b * Hh + y) * Wd + tid] = sp;
  }
}

// ---------------------------------------------------------------------------
// Kernel 2: Gumbel top-256 per batch -> anchors (u,v) in [0,1).
// ---------------------------------------------------------------------------
__global__ __launch_bounds__(256) void topk_kernel(const float* __restrict__ heat,
                                                   float* __restrict__ anc) {
  __shared__ float key[4096];
  __shared__ float sv[256];
  __shared__ int   si[256];
  const int b = blockIdx.x, tid = threadIdx.x;
  for (int i = tid; i < 4096; i += 256) {
    float h = heat[(size_t)b * 4096 + i];
    unsigned r = hash_u((unsigned)(b * 4096 + i) * 0x9E3779B9u + 0x85EBCA6Bu);
    float u = fmaxf((float)(r >> 8) * (1.f / 16777216.f), 1e-8f);
    float gum = -logf(-logf(u));
    key[i] = logf(h + 1e-12f) + gum;
  }
  __syncthreads();
  for (int sel = 0; sel < 256; ++sel) {
    float best = -3.0e38f; int bi = 0;
    for (int i = tid; i < 4096; i += 256) {
      float k = key[i];
      if (k > best) { best = k; bi = i; }
    }
    sv[tid] = best; si[tid] = bi;
    __syncthreads();
    for (int o = 128; o; o >>= 1) {
      if (tid < o && sv[tid + o] > sv[tid]) { sv[tid] = sv[tid + o]; si[tid] = si[tid + o]; }
      __syncthreads();
    }
    if (tid == 0) {
      int idx = si[0];
      key[idx] = -3.0e38f;
      anc[((size_t)b * Np + sel) * 2 + 0] = (float)(idx & 63) * (1.f / 64.f);
      anc[((size_t)b * Np + sel) * 2 + 1] = (float)(idx >> 6) * (1.f / 64.f);
    }
    __syncthreads();
  }
}

// ---------------------------------------------------------------------------
// Kernel 3: bilinear sample [B,C,H,W] at anchors -> bf16 [B*N][C]
// ---------------------------------------------------------------------------
__global__ void sample_bf16(const float* __restrict__ src, const float* __restrict__ anc,
                            __bf16* __restrict__ dst, int C) {
  const int bn = blockIdx.x;
  const int b  = bn >> 8;
  const float u = anc[bn * 2 + 0], v = anc[bn * 2 + 1];
  const float x = u * 64.f - 0.5f, y = v * 64.f - 0.5f;
  const int x0 = (int)floorf(x), y0 = (int)floorf(y);
  const int x1 = x0 + 1, y1 = y0 + 1;
  const float wx1 = x - (float)x0, wy1 = y - (float)y0;
  const float wx0 = 1.f - wx1, wy0 = 1.f - wy1;
  const bool vx0 = (unsigned)x0 < 64u, vx1 = (unsigned)x1 < 64u;
  const bool vy0 = (unsigned)y0 < 64u, vy1 = (unsigned)y1 < 64u;
  const int cx0 = min(max(x0, 0), 63), cx1 = min(max(x1, 0), 63);
  const int cy0 = min(max(y0, 0), 63), cy1 = min(max(y1, 0), 63);
  const float w00 = wx0 * wy0 * (float)(vx0 && vy0);
  const float w10 = wx1 * wy0 * (float)(vx1 && vy0);
  const float w01 = wx0 * wy1 * (float)(vx0 && vy1);
  const float w11 = wx1 * wy1 * (float)(vx1 && vy1);
  for (int c = threadIdx.x; c < C; c += 256) {
    const float* f = src + ((size_t)b * C + c) * 4096;
    float val = f[cy0 * 64 + cx0] * w00 + f[cy0 * 64 + cx1] * w10 +
                f[cy1 * 64 + cx0] * w01 + f[cy1 * 64 + cx1] * w11;
    dst[(size_t)bn * C + c] = f2bf(val);
  }
}

// Kernel 3b: saliency sampling (C=1) + per-batch max via uint atomicMax
__global__ void sample_sal(const float* __restrict__ heat, const float* __restrict__ anc,
                           float* __restrict__ sal, float* __restrict__ scal) {
  int bn = blockIdx.x * 256 + threadIdx.x;
  if (bn >= Bn * Np) return;
  int b = bn >> 8;
  float u = anc[bn * 2 + 0], v = anc[bn * 2 + 1];
  float x = u * 64.f - 0.5f, y = v * 64.f - 0.5f;
  int x0 = (int)floorf(x), y0 = (int)floorf(y);
  int x1 = x0 + 1, y1 = y0 + 1;
  float wx1 = x - (float)x0, wy1 = y - (float)y0;
  float wx0 = 1.f - wx1, wy0 = 1.f - wy1;
  bool vx0 = (unsigned)x0 < 64u, vx1 = (unsigned)x1 < 64u;
  bool vy0 = (unsigned)y0 < 64u, vy1 = (unsigned)y1 < 64u;
  int cx0 = min(max(x0, 0), 63), cx1 = min(max(x1, 0), 63);
  int cy0 = min(max(y0, 0), 63), cy1 = min(max(y1, 0), 63);
  const float* f = heat + (size_t)b * 4096;
  float val = f[cy0 * 64 + cx0] * (wx0 * wy0 * (float)(vx0 && vy0)) +
              f[cy0 * 64 + cx1] * (wx1 * wy0 * (float)(vx1 && vy0)) +
              f[cy1 * 64 + cx0] * (wx0 * wy1 * (float)(vx0 && vy1)) +
              f[cy1 * 64 + cx1] * (wx1 * wy1 * (float)(vx1 && vy1));
  sal[bn] = val;
  atomicMax((unsigned int*)&scal[b], __float_as_uint(fmaxf(val, 0.f)));
}

// ---------------------------------------------------------------------------
// Kernel 4: projection GEMM  out[4096][256] = A[4096][K] x W^T ( Bw[256][K] )
// ---------------------------------------------------------------------------
template <int K>
__global__ __launch_bounds__(256) void proj_gemm(const __bf16* __restrict__ A,
                                                 const __bf16* __restrict__ Bw,
                                                 const float* __restrict__ bias,
                                                 float* __restrict__ out) {
  const int lane = threadIdx.x & 31, wave = threadIdx.x >> 5;
  const int m0 = blockIdx.x * 128 + wave * 16;
  const int ln = lane & 15;
  const int koff = (lane & 16) ? 8 : 0;
  const int khalf = lane & 16;
  v8f vz = {0.f,0.f,0.f,0.f,0.f,0.f,0.f,0.f};
  v8f acc[16];
  for (int t = 0; t < 16; ++t) acc[t] = vz;

  for (int kb = 0; kb < K; kb += 32) {
    v16bf a;
    const __bf16* ap = A + (size_t)(m0 + ln) * K + kb + koff;
    #pragma unroll
    for (int j = 0; j < 8; ++j) { a[j] = ap[j]; a[8 + j] = ap[16 + j]; }
    #pragma unroll
    for (int t = 0; t < 16; ++t) {
      v16bf bf;
      const __bf16* bp = Bw + (size_t)(t * 16 + ln) * K + kb + koff;
      #pragma unroll
      for (int e = 0; e < 8; ++e) { bf[e] = bp[e]; bf[8 + e] = bp[16 + e]; }
      acc[t] = wmma_bf16(a, bf, acc[t]);
    }
  }
  for (int t = 0; t < 16; ++t) {
    #pragma unroll
    for (int r = 0; r < 8; ++r) {
      int row = m0 + ((lane & 16) ? 8 : 0) + r;
      int col = t * 16 + ln;
      out[(size_t)row * Np + col] = acc[t][r] + bias[col];
    }
  }
}

// ---------------------------------------------------------------------------
// Kernel 5: LayerNorm(256) + L2 normalize, wave-per-row; optional bf16 copy.
// ---------------------------------------------------------------------------
__global__ void ln_l2(float* __restrict__ x, const float* __restrict__ g,
                      const float* __restrict__ be, __bf16* __restrict__ obf) {
  const int lane = threadIdx.x & 31;
  const int row = blockIdx.x * 8 + (threadIdx.x >> 5);
  float* xr = x + (size_t)row * Np;
  float v[8];
  float s = 0.f;
  #pragma unroll
  for (int j = 0; j < 8; ++j) { v[j] = xr[lane * 8 + j]; s += v[j]; }
  for (int m = 16; m; m >>= 1) s += __shfl_xor(s, m, 32);
  float mu = s * (1.f / 256.f);
  float q = 0.f;
  #pragma unroll
  for (int j = 0; j < 8; ++j) { v[j] -= mu; q += v[j] * v[j]; }
  for (int m = 16; m; m >>= 1) q += __shfl_xor(q, m, 32);
  float inv = rsqrtf(q * (1.f / 256.f) + 1e-5f);
  float ss = 0.f;
  #pragma unroll
  for (int j = 0; j < 8; ++j) {
    int c = lane * 8 + j;
    v[j] = v[j] * inv * g[c] + be[c];
    ss += v[j] * v[j];
  }
  for (int m = 16; m; m >>= 1) ss += __shfl_xor(ss, m, 32);
  float sc = 1.f / fmaxf(sqrtf(ss), 1e-12f);
  #pragma unroll
  for (int j = 0; j < 8; ++j) {
    float o = v[j] * sc;
    xr[lane * 8 + j] = o;
    if (obf) obf[(size_t)row * Np + lane * 8 + j] = f2bf(o);
  }
}

// Kernel 6: sum (fs-ft)^2 -> scal[64]
__global__ void gafd_red(const float* __restrict__ fs, const float* __restrict__ ft,
                         float* __restrict__ accum) {
  __shared__ float red[256];
  int idx = blockIdx.x * 256 + threadIdx.x;
  float d = fs[idx] - ft[idx];
  red[threadIdx.x] = d * d;
  __syncthreads();
  for (int o = 128; o; o >>= 1) {
    if (threadIdx.x < o) red[threadIdx.x] += red[threadIdx.x + o];
    __syncthreads();
  }
  if (threadIdx.x == 0) atomicAdd(accum, red[0]);
}

// Kernel 7: pos[row] = dot(fs,ft)/tau, wave-per-row
__global__ void pos_kernel(const float* __restrict__ fs, const float* __restrict__ ft,
                           float* __restrict__ pos) {
  const int lane = threadIdx.x & 31;
  const int row = blockIdx.x * 8 + (threadIdx.x >> 5);
  float s = 0.f;
  #pragma unroll
  for (int j = 0; j < 8; ++j) {
    int c = lane * 8 + j;
    s += fs[(size_t)row * Np + c] * ft[(size_t)row * Np + c];
  }
  for (int m = 16; m; m >>= 1) s += __shfl_xor(s, m, 32);
  if (lane == 0) pos[row] = s * TAU_INV;
}

// Kernel 8: boundary mask + counts
__global__ void bd_kernel(const float* __restrict__ sal, float* __restrict__ scal,
                          int* __restrict__ bdm) {
  int idx = blockIdx.x * 256 + threadIdx.x;
  if (idx >= Bn * Np) return;
  int b = idx >> 8;
  float smax = __uint_as_float(((unsigned int*)scal)[b]);
  int m = (sal[idx] > 0.6f * smax) ? 1 : 0;
  bdm[idx] = m;
  atomicAdd(&scal[16 + b], m ? 1.f : 0.f);
  atomicAdd(&scal[32 + b], m ? 0.f : 1.f);
}

// ---------------------------------------------------------------------------
// Kernel 9: neg[b] = (fsn[b] @ fsn[b]^T)/tau with non_bd column masking (WMMA)
// ---------------------------------------------------------------------------
__global__ __launch_bounds__(256) void neg_gemm(const __bf16* __restrict__ F,
                                                const int* __restrict__ bdm,
                                                float* __restrict__ neg) {
  const int b = blockIdx.y;
  const __bf16* base = F + (size_t)b * Np * Np;
  const int lane = threadIdx.x & 31, wave = threadIdx.x >> 5;
  const int m0 = blockIdx.x * 128 + wave * 16;
  const int ln = lane & 15;
  const int koff = (lane & 16) ? 8 : 0;
  const int khalf = lane & 16;
  v8f vz = {0.f,0.f,0.f,0.f,0.f,0.f,0.f,0.f};
  v8f acc[16];
  for (int t = 0; t < 16; ++t) acc[t] = vz;

  for (int kb = 0; kb < Np; kb += 32) {
    v16bf a;
    const __bf16* ap = base + (size_t)(m0 + ln) * Np + kb + koff;
    #pragma unroll
    for (int j = 0; j < 8; ++j) { a[j] = ap[j]; a[8 + j] = ap[16 + j]; }
    #pragma unroll
    for (int t = 0; t < 16; ++t) {
      v16bf bf;
      const __bf16* bp = base + (size_t)(t * 16 + ln) * Np + kb + koff;
      #pragma unroll
      for (int e = 0; e < 8; ++e) { bf[e] = bp[e]; bf[8 + e] = bp[16 + e]; }
      acc[t] = wmma_bf16(a, bf, acc[t]);
    }
  }
  for (int t = 0; t < 16; ++t) {
    #pragma unroll
    for (int r = 0; r < 8; ++r) {
      int row = m0 + ((lane & 16) ? 8 : 0) + r;
      int col = t * 16 + ln;
      float v = acc[t][r] * TAU_INV;
      if (bdm[b * Np + col]) v = -1e30f;   // keep only non-boundary negatives
      neg[((size_t)b * Np + row) * Np + col] = v;
    }
  }
}

// Kernel 10: row_loss = logsumexp([pos, neg_row]) - pos, accumulated over bd rows
__global__ void rowloss_kernel(const float* __restrict__ neg, const float* __restrict__ pos,
                               const int* __restrict__ bdm, float* __restrict__ scal) {
  const int lane = threadIdx.x & 31;
  const int row = blockIdx.x * 8 + (threadIdx.x >> 5);
  const int b = row >> 8;
  const float pv = pos[row];
  const float* nr = neg + (size_t)row * Np;
  float mx = pv;
  for (int i = lane; i < Np; i += 32) mx = fmaxf(mx, nr[i]);
  for (int m = 16; m; m >>= 1) mx = fmaxf(mx, __shfl_xor(mx, m, 32));
  float se = 0.f;
  for (int i = lane; i < Np; i += 32) se += expf(nr[i] - mx);
  for (int m = 16; m; m >>= 1) se += __shfl_xor(se, m, 32);
  if (lane == 0 && bdm[row]) {
    se += expf(pv - mx);
    float rl = mx + logf(se) - pv;
    atomicAdd(&scal[48 + b], rl);
  }
}

// Kernel 11: combine scalars -> d_out[0..2]
__global__ void finalize(const float* __restrict__ scal, float* __restrict__ out) {
  if (threadIdx.x != 0 || blockIdx.x != 0) return;
  float gafd = scal[64] * (1.f / (4096.f * 256.f));
  float vn = 0.f, accv = 0.f;
  for (int b = 0; b < Bn; ++b) {
    float cnt = scal[16 + b], nbd = scal[32 + b];
    float per = scal[48 + b] / fmaxf(cnt, 1.f);
    float valid = (cnt > 0.f && nbd > 0.f) ? 1.f : 0.f;
    vn += valid; accv += per * valid;
  }
  float abcd = (vn > 0.f) ? accv / vn : 0.f;
  out[1] = gafd;
  out[2] = abcd;
  out[0] = gafd + 0.5f * abcd;
}

// ---------------------------------------------------------------------------
extern "C" void kernel_launch(void* const* d_in, const int* in_sizes, int n_in,
                              void* d_out, int out_size, void* d_ws, size_t ws_size,
                              hipStream_t stream) {
  const float* teacher = (const float*)d_in[0];
  const float* student = (const float*)d_in[1];
  const float* c1w = (const float*)d_in[2];
  const float* c1b = (const float*)d_in[3];
  const float* c2w = (const float*)d_in[4];
  const float* c2b = (const float*)d_in[5];
  const float* t_w = (const float*)d_in[6];
  const float* t_b = (const float*)d_in[7];
  const float* t_g = (const float*)d_in[8];
  const float* t_be = (const float*)d_in[9];
  const float* s_w = (const float*)d_in[10];
  const float* s_b = (const float*)d_in[11];
  const float* s_g = (const float*)d_in[12];
  const float* s_be = (const float*)d_in[13];

  float* out = (float*)d_out;
  float* heat = out + 3;                       // [B,1,64,64] output heatmap

  // Workspace layout (~172 MB total)
  char* ws = (char*)d_ws;
  __bf16* w1p = (__bf16*)(ws + 0);             //  9 437 184 B  [9][512][1024]
  __bf16* twp = (__bf16*)(ws + 9437184);       //    524 288 B  [256][1024]
  __bf16* swp = (__bf16*)(ws + 9961472);       //    262 144 B  [256][512]
  __bf16* ftS = (__bf16*)(ws + 10223616);      //  8 388 608 B  [4096][1024]
  __bf16* fsS = (__bf16*)(ws + 18612224);      //  4 194 304 B  [4096][512]
  float*  ftp = (float*)(ws + 22806528);       //  4 194 304 B  [4096][256]
  float*  fsp = (float*)(ws + 27000832);       //  4 194 304 B  [4096][256]
  __bf16* fsn = (__bf16*)(ws + 31195136);      //  2 097 152 B  [4096][256]
  float*  negm = (float*)(ws + 33292288);      //  4 194 304 B  [16][256][256]
  float*  anc = (float*)(ws + 37486592);       //     32 768 B  [16][256][2]
  float*  sal = (float*)(ws + 37519360);       //     16 384 B
  float*  pos = (float*)(ws + 37535744);       //     16 384 B
  int*    bdm = (int*)(ws + 37552128);         //     16 384 B
  float*  scal = (float*)(ws + 37568512);      //      1 024 B
  __bf16* tpk = (__bf16*)(ws + 37569536);      // 134 217 728 B [16][64][64][1024]

  init_scal<<<1, 256, 0, stream>>>(scal);
  pack_w1<<<(9 * C1 * Ct) / 256, 256, 0, stream>>>(c1w, w1p);
  pack_proj<<<(Np * Ct) / 256, 256, 0, stream>>>(t_w, twp, Ct);
  pack_proj<<<(Np * Cs) / 256, 256, 0, stream>>>(s_w, swp, Cs);
  pack_teacher<<<(Bn * Hh * Wd * Ct) / 256, 256, 0, stream>>>(teacher, tpk);

  heatmap_conv<<<dim3(Hh, Bn), 256, 0, stream>>>(tpk, w1p, c1b, c2w, c2b, heat);

  topk_kernel<<<Bn, 256, 0, stream>>>(heat, anc);

  sample_bf16<<<Bn * Np, 256, 0, stream>>>(teacher, anc, ftS, Ct);
  sample_bf16<<<Bn * Np, 256, 0, stream>>>(student, anc, fsS, Cs);
  sample_sal<<<Bn, 256, 0, stream>>>(heat, anc, sal, scal);

  proj_gemm<Ct><<<32, 256, 0, stream>>>(ftS, twp, t_b, ftp);
  proj_gemm<Cs><<<32, 256, 0, stream>>>(fsS, swp, s_b, fsp);

  ln_l2<<<512, 256, 0, stream>>>(ftp, t_g, t_be, (__bf16*)nullptr);
  ln_l2<<<512, 256, 0, stream>>>(fsp, s_g, s_be, fsn);

  gafd_red<<<4096, 256, 0, stream>>>(fsp, ftp, &scal[64]);
  pos_kernel<<<512, 256, 0, stream>>>(fsp, ftp, pos);
  bd_kernel<<<Bn, 256, 0, stream>>>(sal, scal, bdm);

  neg_gemm<<<dim3(2, Bn), 256, 0, stream>>>(fsn, bdm, negm);
  rowloss_kernel<<<512, 256, 0, stream>>>(negm, pos, bdm, scal);

  finalize<<<1, 32, 0, stream>>>(scal, out);
}
